// RGCNConv_spare_78219944394955
// MI455X (gfx1250) — matrix operand order repaired and checked by reading, not compiled, not run on hardware
//
#include <hip/hip_runtime.h>

typedef __attribute__((ext_vector_type(16))) _Float16 v16h;
typedef __attribute__((ext_vector_type(8)))  _Float16 v8h;
typedef __attribute__((ext_vector_type(8)))  float    v8f;
typedef __attribute__((ext_vector_type(8)))  unsigned int v8u;
typedef __attribute__((ext_vector_type(4)))  unsigned int u32x4;
typedef __attribute__((ext_vector_type(8)))  int i32x8;
typedef __attribute__((ext_vector_type(4)))  int i32x4;

#define NN   4096
#define CIN  256
#define COUT 256
#define NREL 8

#ifndef __has_builtin
#define __has_builtin(x) 0
#endif

// ---- compile-time toolchain probes (visible in stderr) ----
#define XSTR_(x) #x
#define SSTR_(x) XSTR_(x)
#pragma message("PROBE clang=" SSTR_(__clang_major__) "." SSTR_(__clang_minor__))
#if __has_builtin(__builtin_amdgcn_tensor_load_to_lds)
#pragma message("PROBE HAS __builtin_amdgcn_tensor_load_to_lds")
#define HAVE_TDM_BUILTIN 1
#else
#pragma message("PROBE NO __builtin_amdgcn_tensor_load_to_lds")
#define HAVE_TDM_BUILTIN 0
#endif
#if __has_builtin(__builtin_amdgcn_s_wait_tensorcnt)
#pragma message("PROBE HAS __builtin_amdgcn_s_wait_tensorcnt")
#else
#pragma message("PROBE NO __builtin_amdgcn_s_wait_tensorcnt")
#endif
#if defined(__has_include)
#if __has_include(<hip/amd_detail/amd_gfx1250_TDM.h>)
#pragma message("PROBE HAS amd_gfx1250_TDM.h")
#else
#pragma message("PROBE NO amd_gfx1250_TDM.h")
#endif
#endif

#define USE_TDM HAVE_TDM_BUILTIN

#define CAT16(a, b) __builtin_shufflevector(a, b, 0,1,2,3,4,5,6,7,8,9,10,11,12,13,14,15)

// ---------------- pack kernels ----------------
__global__ void __launch_bounds__(256)
pack_big(const float* __restrict__ A, const int* __restrict__ E,
         _Float16* __restrict__ Ah, unsigned char* __restrict__ E8) {
    int t = blockIdx.x * 256 + threadIdx.x;
    int base = t * 4;
    float4 a = *(const float4*)(A + base);
    int4   e = *(const int4*)(E + base);
    union { _Float16 h[4]; uint2 u; } ah;
    ah.h[0] = (_Float16)a.x; ah.h[1] = (_Float16)a.y;
    ah.h[2] = (_Float16)a.z; ah.h[3] = (_Float16)a.w;
    *(uint2*)(Ah + base) = ah.u;
    unsigned int eb = (unsigned)(e.x & 255) | ((unsigned)(e.y & 255) << 8) |
                      ((unsigned)(e.z & 255) << 16) | ((unsigned)(e.w & 255) << 24);
    *(unsigned int*)(E8 + base) = eb;
}

__global__ void __launch_bounds__(256)
pack_x(const float* __restrict__ x, _Float16* __restrict__ xh) {
    int t = blockIdx.x * 256 + threadIdx.x;
    int base = t * 4;
    float4 a = *(const float4*)(x + base);
    union { _Float16 h[4]; uint2 u; } hh;
    hh.h[0] = (_Float16)a.x; hh.h[1] = (_Float16)a.y;
    hh.h[2] = (_Float16)a.z; hh.h[3] = (_Float16)a.w;
    *(uint2*)(xh + base) = hh.u;
}

__global__ void __launch_bounds__(256)
pack_wt(const float* __restrict__ W, _Float16* __restrict__ Wth) {
    int t = blockIdx.x * 256 + threadIdx.x;      // t = ((r*256 + c)*256 + k)
    int r = t >> 16;
    int c = (t >> 8) & 255;
    int k = t & 255;
    Wth[t] = (_Float16)W[(r * 256 + k) * 256 + c];
}

// ---------------- Y^T[r][c][j] = sum_k Wt[r][c][k] * x[j][k]  (WMMA) ----------------
__global__ void __launch_bounds__(256)
yt_gemm(const _Float16* __restrict__ xh, const _Float16* __restrict__ Wth,
        _Float16* __restrict__ Yt) {
    int wid  = blockIdx.x * 8 + (threadIdx.x >> 5);
    int r    = wid >> 12;
    int ct   = (wid >> 8) & 15;
    int jt   = wid & 255;
    int lane = threadIdx.x & 31;
    int hi   = (lane >> 4) & 1;
    int l15  = lane & 15;
    const _Float16* wrow = Wth + ((r * 256 + ct * 16 + l15) * 256);
    const _Float16* xrow = xh + ((jt * 16 + l15) * 256);
    v8f acc = {};
#pragma unroll
    for (int k0 = 0; k0 < CIN; k0 += 32) {
        v8h a0 = *(const v8h*)(wrow + k0 + 8 * hi);
        v8h a1 = *(const v8h*)(wrow + k0 + 16 + 8 * hi);
        v8h b0 = *(const v8h*)(xrow + k0 + 8 * hi);
        v8h b1 = *(const v8h*)(xrow + k0 + 16 + 8 * hi);
        v16h af = CAT16(a0, a1);
        v16h bf = CAT16(b0, b1);
        acc = __builtin_amdgcn_wmma_f32_16x16x32_f16(false, af, false, bf,
                                                     (short)0, acc, false, false);
    }
#pragma unroll
    for (int v = 0; v < 8; ++v) {
        int row = ct * 16 + v + 8 * hi;
        Yt[(r * 256 + row) * 4096 + jt * 16 + l15] = (_Float16)acc[v];
    }
}

#if USE_TDM
__device__ __forceinline__ void wait_tensorcnt(int n) {
#if __has_builtin(__builtin_amdgcn_s_wait_tensorcnt)
    if (n == 0) __builtin_amdgcn_s_wait_tensorcnt(0);
    else        __builtin_amdgcn_s_wait_tensorcnt(1);
#else
    if (n == 0) asm volatile("s_wait_tensorcnt 0x0" ::: "memory");
    else        asm volatile("s_wait_tensorcnt 0x1" ::: "memory");
#endif
}

// 2D TDM: tile = 32 halves (contiguous) x 64 rows, row stride 4096 halves.
__device__ __forceinline__ void tdm_load_slab(const _Float16* gbase, unsigned lds_off) {
    unsigned long long ga = (unsigned long long)(size_t)gbase;
    u32x4 g0;
    g0[0] = 1u;                                            // count=1, user descriptor
    g0[1] = lds_off;                                       // lds_addr (bytes)
    g0[2] = (unsigned)(ga & 0xFFFFFFFFull);                // global_addr[31:0]
    g0[3] = (unsigned)((ga >> 32) & 0x01FFFFFFull) | (2u << 30);  // [56:32] | type=2
    i32x8 g1;
    g1[0] = (int)(1u << 16);                               // wg_mask=0, data_size=2B
    g1[1] = (int)((NN & 0xFFFFu) << 16);                   // tensor_dim0[15:0]=4096
    g1[2] = (int)((NN >> 16) | (256u << 16));              // dim0[31:16] | tensor_dim1=256
    g1[3] = (int)(32u << 16);                              // tile_dim0=32
    g1[4] = (int)64u;                                      // tile_dim1=64, tile_dim2=0
    g1[5] = (int)NN;                                       // tensor_dim0_stride=4096
    g1[6] = 0;
    g1[7] = 0;
    i32x4 g2; g2[0] = 1; g2[1] = 1; g2[2] = 0; g2[3] = 0;  // tensor_dim2=1, tensor_dim3=1
    i32x4 g3; g3[0] = 0; g3[1] = (int)(1u << 16); g3[2] = 0; g3[3] = 0; // tensor_dim4=1
#if __clang_major__ >= 23
    i32x8 gz = {};   // therock/clang-23 6-arg form (extra group, zero-filled)
    __builtin_amdgcn_tensor_load_to_lds(g0, g1, g2, g3, gz, 0);
#else
    __builtin_amdgcn_tensor_load_to_lds(g0, g1, g2, g3, 0);
#endif
}
#endif

// ------------- main: out = sum_r (A .* [E==r]) @ Y_r + bias -------------
__global__ void __launch_bounds__(128)
rgcn_main(const _Float16* __restrict__ Ah, const unsigned char* __restrict__ E8,
          const _Float16* __restrict__ Yt, const float* __restrict__ bias,
          float* __restrict__ out) {
    __shared__ _Float16 slab[2][NREL * 64 * 32];   // 2 x 32 KB: [buf][r][n][k]
    int t    = threadIdx.x;
    int wave = t >> 5;
    int lane = t & 31;
    int hi   = (lane >> 4) & 1;
    int l15  = lane & 15;
    int i0   = blockIdx.x * 64 + wave * 16;
    int c0   = blockIdx.y * 64;
    int irow = i0 + l15;
    const _Float16*      arow = Ah + (irow * 4096);
    const unsigned char* erow = E8 + (irow * 4096);

    v8f acc[4] = {{}, {}, {}, {}};

#if USE_TDM
    unsigned lds_base = (unsigned)(size_t)(&slab[0][0]);
    if (wave == 0) {
#pragma unroll
        for (int r = 0; r < NREL; ++r)
            tdm_load_slab(Yt + ((r * 256 + c0) * 4096 + 0),
                          lds_base + (unsigned)(r * 64 * 32 * 2));
    }
#endif

    for (int k0 = 0; k0 < NN; k0 += 32) {
        int buf = (k0 >> 5) & 1;
        __syncthreads();                 // all waves done reading slab[buf^1]
#if USE_TDM
        if (wave == 0) {
            if (k0 + 32 < NN) {
#pragma unroll
                for (int r = 0; r < NREL; ++r)
                    tdm_load_slab(Yt + ((r * 256 + c0) * 4096 + (k0 + 32)),
                                  lds_base + (unsigned)(((buf ^ 1) * NREL + r) * 64 * 32 * 2));
                wait_tensorcnt(1);       // current slab done, prefetch in flight
            } else {
                wait_tensorcnt(0);
            }
        }
#else
        // cooperative stage of current slab: 2048 x 16B chunks, 16 per thread
#pragma unroll
        for (int q = 0; q < 16; ++q) {
            int idx = q * 128 + t;
            int rn  = idx >> 2;
            int ch  = idx & 3;
            int r   = rn >> 6;
            int n   = rn & 63;
            const uint4* g = (const uint4*)(Yt + ((r * 256 + c0 + n) * 4096 + k0 + ch * 8));
            *(uint4*)(&slab[buf][(r * 64 + n) * 32 + ch * 8]) = *g;
        }
#endif
        __syncthreads();                 // slab[buf] ready for all waves

        if (k0 + 32 < NN) {              // stream hint for next A/E fragments
            __builtin_prefetch(arow + k0 + 32, 0, 3);
            __builtin_prefetch(erow + k0 + 32, 0, 3);
        }

        // A fragment (f16) + edge fragment packed as u16 pairs matching A's VGPR halves
        v8h a0 = *(const v8h*)(arow + k0 + 8 * hi);
        v8h a1 = *(const v8h*)(arow + k0 + 16 + 8 * hi);
        v8u aw = (v8u)CAT16(a0, a1);
        uint2 e0 = *(const uint2*)(erow + k0 + 8 * hi);
        uint2 e1 = *(const uint2*)(erow + k0 + 16 + 8 * hi);
        unsigned int ew[8];
#pragma unroll
        for (int w = 0; w < 8; ++w) {
            unsigned int d = (w < 2) ? e0.x : (w < 4) ? e0.y : (w < 6) ? e1.x : e1.y;
            int sh = (w & 1) * 16;
            ew[w] = ((d >> sh) & 0xFFu) | (((d >> (sh + 8)) & 0xFFu) << 16);
        }

#pragma unroll
        for (int r = 0; r < NREL; ++r) {
            unsigned int rr = (unsigned)r * 0x00010001u;
            v8u mw;
#pragma unroll
            for (int w = 0; w < 8; ++w) {
                unsigned int tt  = ew[w] ^ rr;
                unsigned int qq  = tt + 0x000F000Fu;
                unsigned int eqb = ((~qq) >> 4) & 0x00010001u;
                unsigned int m   = (eqb << 16) - eqb;
                mw[w] = aw[w] & m;
            }
            v16h ma = (v16h)mw;
#pragma unroll
            for (int nt = 0; nt < 4; ++nt) {
                const v8h* lp = (const v8h*)(&slab[buf][(r * 64 + nt * 16 + l15) * 32]);
                v16h bf = CAT16(lp[hi], lp[2 + hi]);
                acc[nt] = __builtin_amdgcn_wmma_f32_16x16x32_f16(false, ma, false, bf,
                                                                 (short)0, acc[nt],
                                                                 false, false);
            }
        }
    }

#pragma unroll
    for (int nt = 0; nt < 4; ++nt) {
        int cc = c0 + nt * 16 + l15;
        float bv = bias[cc];
#pragma unroll
        for (int v = 0; v < 8; ++v) {
            int rowi = i0 + v + 8 * hi;
            out[rowi * 256 + cc] = acc[nt][v] + bv;
        }
    }
}

extern "C" void kernel_launch(void* const* d_in, const int* in_sizes, int n_in,
                              void* d_out, int out_size, void* d_ws, size_t ws_size,
                              hipStream_t stream) {
    const float* x    = (const float*)d_in[0];
    const float* A    = (const float*)d_in[1];
    const int*   E    = (const int*)d_in[2];
    const float* W    = (const float*)d_in[3];
    const float* bias = (const float*)d_in[4];
    float* out = (float*)d_out;

    char* ws = (char*)d_ws;
    _Float16* xh  = (_Float16*)(ws);                    //  2 MB  [4096][256]
    _Float16* Wth = (_Float16*)(ws + (2ull  << 20));    //  1 MB  [8][256][256]
    _Float16* Yt  = (_Float16*)(ws + (3ull  << 20));    // 16 MB  [8][256][4096]
    _Float16* Ah  = (_Float16*)(ws + (19ull << 20));    // 32 MB  [4096][4096]
    unsigned char* E8 = (unsigned char*)(ws + (51ull << 20)); // 16 MB

    pack_big<<<(NN * NN / 4) / 256, 256, 0, stream>>>(A, E, Ah, E8);
    pack_x<<<(NN * CIN / 4) / 256, 256, 0, stream>>>(x, xh);
    pack_wt<<<(NREL * CIN * COUT) / 256, 256, 0, stream>>>(W, Wth);
    yt_gemm<<<4096, 256, 0, stream>>>(xh, Wth, Yt);
    rgcn_main<<<dim3(64, 4), 128, 0, stream>>>(Ah, E8, Yt, bias, out);
}
